// LSTMModel_19988777796259
// MI455X (gfx1250) — matrix operand order repaired and checked by reading, not compiled, not run on hardware
//
#include <hip/hip_runtime.h>
#include <hip/hip_bf16.h>

// ---------------------------------------------------------------------------
// Problem constants (from reference setup_inputs)
// ---------------------------------------------------------------------------
#define BB   64      // batch
#define TT   1024    // timesteps
#define FF   256     // features
#define HH   1024    // hidden
#define GH   4096    // 4*H
#define KC   1280    // combined K = H + F
#define OO   256     // output
#define NWG  64      // persistent workgroups (co-resident)

typedef __bf16 bf16_t;
typedef bf16_t v16bf __attribute__((ext_vector_type(16)));
typedef bf16_t v8bf  __attribute__((ext_vector_type(8)));
typedef float  v8f   __attribute__((ext_vector_type(8)));

// sigma=1, radius=4 normalized Gaussian taps
__constant__ float GW[9] = {
    1.3383062e-4f, 4.4318616e-3f, 5.3991128e-2f, 2.4197157e-1f,
    3.9894360e-1f, 2.4197157e-1f, 5.3991128e-2f, 4.4318616e-3f, 1.3383062e-4f};

// ---------------------------------------------------------------------------
// Blur pass 1: along F (contiguous axis), edge clamp.  x[B,T,F] -> tmp[B,T,F]
// ---------------------------------------------------------------------------
__global__ __launch_bounds__(256) void blur_f_kernel(
    const float* __restrict__ x, float* __restrict__ tmp) {
  int idx  = blockIdx.x * 256 + threadIdx.x;      // B*T*F = 2^24
  int f    = idx & (FF - 1);
  int base = idx - f;
  float s = 0.f;
#pragma unroll
  for (int k = 0; k < 9; ++k) {
    int fc = f + k - 4;
    fc = fc < 0 ? 0 : (fc > FF - 1 ? FF - 1 : fc);
    s += GW[k] * x[base + fc];
  }
  tmp[idx] = s;
}

// ---------------------------------------------------------------------------
// Blur pass 2: along T, edge clamp; emits bf16 in [T][B][F] layout so each
// timestep's A-matrix slab (64x256) is a contiguous 32KB block.
// ---------------------------------------------------------------------------
__global__ __launch_bounds__(256) void blur_t_kernel(
    const float* __restrict__ tmp, bf16_t* __restrict__ xb) {
  int idx = blockIdx.x * 256 + threadIdx.x;
  int f = idx & (FF - 1);
  int t = (idx >> 8) & (TT - 1);
  int b = idx >> 18;
  float s = 0.f;
#pragma unroll
  for (int k = 0; k < 9; ++k) {
    int tc = t + k - 4;
    tc = tc < 0 ? 0 : (tc > TT - 1 ? TT - 1 : tc);
    s += GW[k] * tmp[((b << 10) + tc) * FF + f];
  }
  xb[((t << 6) + b) * FF + f] = (bf16_t)s;
}

// ---------------------------------------------------------------------------
// Weight fuse+convert: Wc[g][k] = (k<H ? W_hh[g][k] : W_ih[g][k-H]) as bf16.
// Row-major [4096 x 1280]: each output column g is one contiguous K-row.
// ---------------------------------------------------------------------------
__global__ __launch_bounds__(256) void wc_convert_kernel(
    const float* __restrict__ W_ih, const float* __restrict__ W_hh,
    bf16_t* __restrict__ Wc) {
  int idx = blockIdx.x * 256 + threadIdx.x;       // 4096*1280
  int g = idx / KC;
  int k = idx - g * KC;
  float v = (k < HH) ? W_hh[g * HH + k] : W_ih[g * FF + (k - HH)];
  Wc[idx] = (bf16_t)v;
}

// W_fc [256 x 1024] -> bf16, and reset the grid-sync flags (ws is poisoned).
__global__ __launch_bounds__(256) void wfc_init_kernel(
    const float* __restrict__ W_fc, bf16_t* __restrict__ Wfc, int* sync) {
  int idx = blockIdx.x * 256 + threadIdx.x;       // 262144
  Wfc[idx] = (bf16_t)W_fc[idx];
  if (idx == 0) { sync[0] = 0; sync[1] = 0; }
}

// ---------------------------------------------------------------------------
// Device-wide barrier for the persistent kernel (64 co-resident WGs).
// ---------------------------------------------------------------------------
__device__ __forceinline__ void grid_sync(int* cnt, volatile int* gen) {
  __threadfence();
  __syncthreads();
  if (threadIdx.x == 0) {
    int g = *gen;
    if (atomicAdd(cnt, 1) == NWG - 1) {
      *cnt = 0;
      __threadfence();
      atomicAdd((int*)gen, 1);
    } else {
      while (*gen == g) __builtin_amdgcn_s_sleep(1);
    }
  }
  __syncthreads();
  __threadfence();
}

// ---------------------------------------------------------------------------
// WMMA fragment layouts as loaded here (wave32); A and B share one K order.
// A (16x32 bf16): lane<16 holds row M, K = k0..k0+15 contiguous;
//                 lane>=16 holds row M, K = k0+16..k0+31.
// B (32x16 bf16): lane<16 holds col N, K = k0..k0+15 contiguous;
//                 lane>=16 holds col N, K = k0+16..k0+31.
// ---------------------------------------------------------------------------
union AFrag { v16bf v; v8bf h[2]; };

// A fragment: base = uniform scalar pointer, aoff = per-lane byte offset of
// (row, K-half); 16 contiguous K values per half-lane (one 32B load).
__device__ __forceinline__ v16bf load_a(const char* base, int aoff, int k0) {
  return *(const v16bf*)(base + aoff + (k0 << 1));
}

// B fragment: 16 contiguous K values per half-lane (one 32B vector load).
__device__ __forceinline__ v16bf load_b(const char* base, int boff, int k0) {
  return *(const v16bf*)(base + boff + (k0 << 1));
}

#define WMMA_BF16(A, Bv, C) \
  __builtin_amdgcn_wmma_f32_16x16x32_bf16(false, (A), false, (Bv), (short)0, (C), false, false)

// ---------------------------------------------------------------------------
// Persistent LSTM kernel.
// WG n owns gate columns { n*16 + j + q*1024 : j in [0,16), q in [0,4) }.
// Per step: 64x64 GEMM tile (K=1280 fused [h;x]) via v_wmma_f32_16x16x32_bf16
// into LDS -> gate nonlinearity with c in registers -> write 16 h-columns ->
// one grid barrier.  h is ping-pong double buffered across the barrier.
// ---------------------------------------------------------------------------
__global__ __launch_bounds__(256, 1) void lstm_persistent_kernel(
    const bf16_t* __restrict__ xb,  const bf16_t* __restrict__ Wc,
    const bf16_t* __restrict__ Wfc,
    const float* __restrict__ b_ih, const float* __restrict__ b_hh,
    const float* __restrict__ b_fc,
    bf16_t* hb0, bf16_t* hb1, float* __restrict__ out, int* sync) {
  __shared__ float Gt[64][68];                    // 64x64 G-tile (+pad)

  const int tid  = threadIdx.x;
  const int wg   = blockIdx.x;                    // 0..63
  const int wave = tid >> 5;                      // 8 waves
  const int lane = tid & 31;
  const int half = lane >> 4;
  const int lr   = lane & 15;

  const int rblk = (wave & 3) << 4;               // A row block: 0/16/32/48
  const int q0   = wave >> 2;                     // this wave: strips q0, q0+2
  const int n0   = wg << 4;                       // base hidden column

  // zero h ping buffer (16384 threads x 4 bf16 = 64x1024)
  {
    uint2 z; z.x = 0u; z.y = 0u;
    *((uint2*)(hb0 + (size_t)(wg * 256 + tid) * 4)) = z;
  }

  // Per-lane byte offsets (computed once; all hot loads are base+int32).
  const int aoff_h = ((rblk + lr) * HH + (half << 4)) * 2;  // into h buffer
  const int aoff_x = ((rblk + lr) * FF + (half << 4)) * 2;  // into x slab
  const int boff0  = ((q0 * HH + n0 + lr) * KC + (half << 4)) * 2;
  const int boff1  = (((q0 + 2) * HH + n0 + lr) * KC + (half << 4)) * 2;
  const char* wc_c = (const char*)Wc;

  // phase-2 element ownership: thread -> (batch pb, 4 consecutive j)
  const int pb = tid & 63;
  const int pj = (tid >> 6) << 2;
  float c[4] = {0.f, 0.f, 0.f, 0.f};
  float bias[4][4];
#pragma unroll
  for (int q = 0; q < 4; ++q)
#pragma unroll
    for (int e = 0; e < 4; ++e) {
      int g = q * HH + n0 + pj + e;
      bias[q][e] = b_ih[g] + b_hh[g];
    }

  grid_sync(sync, sync + 1);

  bf16_t* hr = hb0;                               // read h (step t-1)
  bf16_t* hw = hb1;                               // write h (step t)

  for (int t = 0; t < TT; ++t) {
    const char* hc = (const char*)hr;
    const char* xc = (const char*)(xb + (size_t)t * (BB * FF));

    // ---- phase 1: G[64 x 64] = [h ; x_t] (64x1280) * Wc-tile (1280x64) ----
    v8f acc0 = {};
    v8f acc1 = {};
    // K in [0, 1024): A rows come from h   (branch-free inner loop)
#pragma unroll 4
    for (int k0 = 0; k0 < HH; k0 += 32) {
      v16bf a   = load_a(hc, aoff_h, k0);
      v16bf bv0 = load_b(wc_c, boff0, k0);
      v16bf bv1 = load_b(wc_c, boff1, k0);
      acc0 = WMMA_BF16(a, bv0, acc0);
      acc1 = WMMA_BF16(a, bv1, acc1);
    }
    // K in [1024, 1280): A rows come from x_t
#pragma unroll
    for (int k0 = 0; k0 < FF; k0 += 32) {
      v16bf a   = load_a(xc, aoff_x, k0);
      v16bf bv0 = load_b(wc_c, boff0, HH + k0);
      v16bf bv1 = load_b(wc_c, boff1, HH + k0);
      acc0 = WMMA_BF16(a, bv0, acc0);
      acc1 = WMMA_BF16(a, bv1, acc1);
    }
    // D layout: lane<16 -> M = v, N = lane; lane>=16 -> M = 8+v, N = lane-16
    {
      int row = rblk + (half << 3);
#pragma unroll
      for (int v = 0; v < 8; ++v) {
        Gt[row + v][(q0 << 4) + lr]       = acc0[v];
        Gt[row + v][((q0 + 2) << 4) + lr] = acc1[v];
      }
    }
    __syncthreads();

    // ---- phase 2: gates (strip q holds gate q: i,f,g,o), c in registers ----
#pragma unroll
    for (int e = 0; e < 4; ++e) {
      int j = pj + e;
      float gi = Gt[pb][j]      + bias[0][e];
      float gf = Gt[pb][16 + j] + bias[1][e];
      float gg = Gt[pb][32 + j] + bias[2][e];
      float go = Gt[pb][48 + j] + bias[3][e];
      float si = 1.f / (1.f + __expf(-gi));
      float sf = 1.f / (1.f + __expf(-gf));
      float tg = tanhf(gg);
      float so = 1.f / (1.f + __expf(-go));
      float cc = sf * c[e] + si * tg;
      c[e] = cc;
      hw[(size_t)pb * HH + n0 + j] = (bf16_t)(so * tanhf(cc));
    }

    grid_sync(sync, sync + 1);                    // publish h to all WGPs
    bf16_t* tp = hr; hr = hw; hw = tp;            // ping-pong
  }

  // ---- final FC: out[64 x 256] = h_last (64x1024) @ W_fc^T + b_fc ----
  if (wg < 4) {
    const int nfc = wg << 6;
    const int c0  = nfc + (q0 << 4) + lr;
    const int c1  = nfc + ((q0 + 2) << 4) + lr;
    const int fb0 = (c0 * HH + (half << 4)) * 2;
    const int fb1 = (c1 * HH + (half << 4)) * 2;
    const char* fc_c = (const char*)Wfc;
    const char* hc   = (const char*)hr;
    v8f acc0 = {};
    v8f acc1 = {};
#pragma unroll 4
    for (int k0 = 0; k0 < HH; k0 += 32) {
      v16bf a   = load_a(hc, aoff_h, k0);
      v16bf bv0 = load_b(fc_c, fb0, k0);
      v16bf bv1 = load_b(fc_c, fb1, k0);
      acc0 = WMMA_BF16(a, bv0, acc0);
      acc1 = WMMA_BF16(a, bv1, acc1);
    }
    int row = rblk + (half << 3);
#pragma unroll
    for (int v = 0; v < 8; ++v) {
      out[(row + v) * OO + c0] = acc0[v] + b_fc[c0];
      out[(row + v) * OO + c1] = acc1[v] + b_fc[c1];
    }
  }
}

// ---------------------------------------------------------------------------
// Host-side orchestration (graph-capture safe: only kernel launches).
// ---------------------------------------------------------------------------
extern "C" void kernel_launch(void* const* d_in, const int* in_sizes, int n_in,
                              void* d_out, int out_size, void* d_ws, size_t ws_size,
                              hipStream_t stream) {
  const float* x    = (const float*)d_in[0];
  const float* W_ih = (const float*)d_in[1];
  const float* W_hh = (const float*)d_in[2];
  const float* b_ih = (const float*)d_in[3];
  const float* b_hh = (const float*)d_in[4];
  const float* W_fc = (const float*)d_in[5];
  const float* b_fc = (const float*)d_in[6];
  float* out = (float*)d_out;

  char* ws = (char*)d_ws;
  const size_t SZ_TMP = (size_t)BB * TT * FF * 4;   // 64 MB fp32 blur temp
  const size_t SZ_XB  = (size_t)BB * TT * FF * 2;   // 32 MB bf16 [T][B][F]
  const size_t SZ_WC  = (size_t)GH * KC * 2;        // 10 MB fused weights
  const size_t SZ_WFC = (size_t)OO * HH * 2;        // 0.5 MB
  const size_t SZ_H   = (size_t)BB * HH * 2;        // 128 KB per h buffer

  float*  tmp = (float*)(ws);
  bf16_t* xbp = (bf16_t*)(ws + SZ_TMP);
  bf16_t* Wc  = (bf16_t*)(ws + SZ_TMP + SZ_XB);
  bf16_t* Wfc = (bf16_t*)(ws + SZ_TMP + SZ_XB + SZ_WC);
  bf16_t* hb0 = (bf16_t*)(ws + SZ_TMP + SZ_XB + SZ_WC + SZ_WFC);
  bf16_t* hb1 = (bf16_t*)(ws + SZ_TMP + SZ_XB + SZ_WC + SZ_WFC + SZ_H);
  int*   sync = (int*)  (ws + SZ_TMP + SZ_XB + SZ_WC + SZ_WFC + 2 * SZ_H);

  const int elems = BB * TT * FF;                   // 2^24
  blur_f_kernel<<<elems / 256, 256, 0, stream>>>(x, tmp);
  blur_t_kernel<<<elems / 256, 256, 0, stream>>>(tmp, xbp);
  wc_convert_kernel<<<(GH * KC) / 256, 256, 0, stream>>>(W_ih, W_hh, Wc);
  wfc_init_kernel<<<(OO * HH) / 256, 256, 0, stream>>>(W_fc, Wfc, sync);
  lstm_persistent_kernel<<<NWG, 256, 0, stream>>>(
      xbp, Wc, Wfc, b_ih, b_hh, b_fc, hb0, hb1, out, sync);
}
